// RL_154618823097
// MI455X (gfx1250) — matrix operand order repaired and checked by reading, not compiled, not run on hardware
//
#include <hip/hip_runtime.h>
#include <hip/hip_bf16.h>
#include <math.h>

// ---------------- constants ----------------
#define Bsz 4
#define Lsz 4096
#define Dsz 1024
#define Hh  16
#define DKV 64
#define KSZ 4

// 1 = use gfx1250 async vmem->LDS copies (ASYNCcnt) for GEMM tile staging
#define USE_ASYNC_COPY 1

typedef __bf16 bf16_t;
typedef bf16_t v16bf __attribute__((ext_vector_type(16)));
typedef float  v8f   __attribute__((ext_vector_type(8)));
typedef unsigned int u32x8 __attribute__((ext_vector_type(8)));

__device__ __forceinline__ unsigned short f2bf(float f) {
    unsigned int u = __float_as_uint(f);
    unsigned int r = u + 0x7FFFu + ((u >> 16) & 1u);   // round-to-nearest-even
    return (unsigned short)(r >> 16);
}

#if USE_ASYNC_COPY
// gfx1250 async global->LDS b128 copy: VDST = per-lane LDS byte offset, ADDR = 64-bit global addr.
__device__ __forceinline__ void async_ld_b128(unsigned lds_byte_off, const void* gptr) {
    asm volatile("global_load_async_to_lds_b128 %0, %1, off"
                 :: "v"(lds_byte_off), "v"(gptr) : "memory");
}
__device__ __forceinline__ void wait_async0() {
    asm volatile("s_wait_asynccnt 0" ::: "memory");
}
#endif

// ---------------- f32 -> bf16 elementwise ----------------
__global__ __launch_bounds__(256) void cvt_bf16_kernel(const float* __restrict__ in,
                                                       unsigned short* __restrict__ out,
                                                       size_t n) {
    size_t i = ((size_t)blockIdx.x * 256 + threadIdx.x) * 4;
    if (i + 3 < n) {
        float4 v = *(const float4*)(in + i);
        out[i + 0] = f2bf(v.x);
        out[i + 1] = f2bf(v.y);
        out[i + 2] = f2bf(v.z);
        out[i + 3] = f2bf(v.w);
    }
}

// ---------------- transpose + convert: WT[n*K + k] = bf16(W[k*C + n]) ----------------
__global__ __launch_bounds__(1024) void transpose_cvt_kernel(const float* __restrict__ W,
                                                             unsigned short* __restrict__ WT,
                                                             int R, int C) {
    __shared__ float tile[32][33];
    int x = blockIdx.x * 32 + threadIdx.x;   // col (n)
    int y = blockIdx.y * 32 + threadIdx.y;   // row (k)
    tile[threadIdx.y][threadIdx.x] = W[(size_t)y * C + x];
    __syncthreads();
    int tk = blockIdx.y * 32 + threadIdx.x;  // k
    int tn = blockIdx.x * 32 + threadIdx.y;  // n
    WT[(size_t)tn * R + tk] = f2bf(tile[threadIdx.x][threadIdx.y]);
}

// ---------------- bf16 WMMA GEMM:  C[M,N] = A[M,K] * B[K,N],  B given transposed (N x K) ----------------
// block = 256 threads (8 waves, wave32). Block tile 128x128, K-step 32, double-buffered LDS.
// Wave grid 2(M) x 4(N): each wave computes 64x32 = 4x2 WMMA 16x16 tiles.
#define ASTR 40                     // 32 + 8 ushort pad, keeps 16B alignment (80B row)
#define TILE_USH (128 * ASTR)       // ushorts per tile buffer
#define TILE_BYTES (TILE_USH * 2)   // 10240 B

__global__ __launch_bounds__(256) void gemm_bf16_wmma(const unsigned short* __restrict__ A,
                                                      const unsigned short* __restrict__ BT,
                                                      float* __restrict__ C,
                                                      int M, int N, int K) {
    __shared__ __align__(16) unsigned short As[2][TILE_USH];
    __shared__ __align__(16) unsigned short Bs[2][TILE_USH];

    const int tid  = threadIdx.x;
    const int lane = tid & 31;
    const int wave = tid >> 5;
    const int wm   = (wave >> 2) * 64;
    const int wn   = (wave & 3) * 32;
    const int m0   = blockIdx.y * 128;
    const int n0   = blockIdx.x * 128;

    const int l15 = lane & 15;
    const int lhi = lane >> 4;

    const int lrow  = tid >> 1;          // 0..127
    const int lpart = (tid & 1) * 16;    // 0 / 16

    v8f acc[4][2];
    const v8f zero = {0.f, 0.f, 0.f, 0.f, 0.f, 0.f, 0.f, 0.f};
#pragma unroll
    for (int i = 0; i < 4; ++i)
#pragma unroll
        for (int j = 0; j < 2; ++j) acc[i][j] = zero;

    const unsigned short* aRow = A  + (size_t)(m0 + lrow) * K + lpart;
    const unsigned short* bRow = BT + (size_t)(n0 + lrow) * K + lpart;

#if USE_ASYNC_COPY
    const unsigned ldsA0 = (unsigned)(uintptr_t)&As[0][lrow * ASTR + lpart];
    const unsigned ldsB0 = (unsigned)(uintptr_t)&Bs[0][lrow * ASTR + lpart];
    // stage tile 0 into buffer 0 via the async copy engine
    async_ld_b128(ldsA0, aRow);
    async_ld_b128(ldsA0 + 16u, aRow + 8);
    async_ld_b128(ldsB0, bRow);
    async_ld_b128(ldsB0 + 16u, bRow + 8);
    wait_async0();
    __syncthreads();
#else
    *(uint4*)&As[0][lrow * ASTR + lpart] = *(const uint4*)(aRow);
    *(uint4*)&Bs[0][lrow * ASTR + lpart] = *(const uint4*)(bRow);
    __syncthreads();
#endif

    int buf = 0;
    for (int k0 = 0; k0 < K; k0 += 32) {
        const bool more = (k0 + 32) < K;
        // ---- kick off prefetch of tile k0+32 into the other buffer ----
#if USE_ASYNC_COPY
        if (more) {
            async_ld_b128(ldsA0 + (unsigned)((buf ^ 1) * TILE_BYTES), aRow + k0 + 32);
            async_ld_b128(ldsA0 + 16u + (unsigned)((buf ^ 1) * TILE_BYTES), aRow + k0 + 40);
            async_ld_b128(ldsB0 + (unsigned)((buf ^ 1) * TILE_BYTES), bRow + k0 + 32);
            async_ld_b128(ldsB0 + 16u + (unsigned)((buf ^ 1) * TILE_BYTES), bRow + k0 + 40);
        }
#else
        uint4 na, nb;
        if (more) {
            na = *(const uint4*)(aRow + k0 + 32);
            nb = *(const uint4*)(bRow + k0 + 32);
        }
#endif
        // ---- compute on current buffer ----
        v16bf afr[4], bfr[2];
#pragma unroll
        for (int i = 0; i < 4; ++i) {
            // A 16x32 bf16 layout: lane&15 = row, lane>>4 selects K-halves {0..7,16..23} vs {8..15,24..31}
            const unsigned short* p = &As[buf][(wm + i * 16 + l15) * ASTR + lhi * 8];
            uint4 x0 = *(const uint4*)p;
            uint4 x1 = *(const uint4*)(p + 16);
            u32x8 t;
            t[0] = x0.x; t[1] = x0.y; t[2] = x0.z; t[3] = x0.w;
            t[4] = x1.x; t[5] = x1.y; t[6] = x1.z; t[7] = x1.w;
            afr[i] = __builtin_bit_cast(v16bf, t);
        }
#pragma unroll
        for (int j = 0; j < 2; ++j) {
            // B 32x16 bf16 layout: lane&15 = col, lane>>4 selects K-halves {0..7,16..23} vs {8..15,24..31}
            const unsigned short* p = &Bs[buf][(wn + j * 16 + l15) * ASTR + lhi * 8];
            uint4 x0 = *(const uint4*)p;
            uint4 x1 = *(const uint4*)(p + 16);
            u32x8 t;
            t[0] = x0.x; t[1] = x0.y; t[2] = x0.z; t[3] = x0.w;
            t[4] = x1.x; t[5] = x1.y; t[6] = x1.z; t[7] = x1.w;
            bfr[j] = __builtin_bit_cast(v16bf, t);
        }
#pragma unroll
        for (int i = 0; i < 4; ++i)
#pragma unroll
            for (int j = 0; j < 2; ++j)
                acc[i][j] = __builtin_amdgcn_wmma_f32_16x16x32_bf16(
                    false, afr[i], false, bfr[j], (short)0, acc[i][j], false, false);

        // ---- publish the prefetched buffer ----
        if (more) {
#if USE_ASYNC_COPY
            wait_async0();
#else
            *(uint4*)&As[buf ^ 1][lrow * ASTR + lpart] = na;
            *(uint4*)&Bs[buf ^ 1][lrow * ASTR + lpart] = nb;
#endif
            __syncthreads();
            buf ^= 1;
        }
    }

    // C/D layout: VGPR r -> M = r (lanes 0-15) / r+8 (lanes 16-31); N = lane&15
#pragma unroll
    for (int i = 0; i < 4; ++i)
#pragma unroll
        for (int j = 0; j < 2; ++j) {
            int col  = n0 + wn + j * 16 + l15;
            int mrow = m0 + wm + i * 16 + lhi * 8;
#pragma unroll
            for (int r = 0; r < 8; ++r)
                C[(size_t)(mrow + r) * N + col] = acc[i][j][r];
        }
}

// ---------------- lr = softplus(x @ Wlr + BASE_LR), one block per (b,l) row ----------------
__global__ __launch_bounds__(256) void lr_kernel(const float* __restrict__ x,
                                                 const float* __restrict__ Wlr,
                                                 float* __restrict__ lr) {
    __shared__ float sx[Dsz];
    __shared__ float part[256];
    const int tid = threadIdx.x;
    const size_t row = blockIdx.x;
    for (int i = tid; i < Dsz; i += 256) sx[i] = x[row * Dsz + i];
    __syncthreads();
    const int h = tid & 15;
    const int chunk = tid >> 4;          // 16 chunks of 64
    float p = 0.f;
    for (int d = chunk * 64; d < chunk * 64 + 64; ++d) p += sx[d] * Wlr[d * Hh + h];
    part[tid] = p;
    __syncthreads();
    if (tid < 16) {
        float s = 0.001f;
        for (int c = 0; c < 16; ++c) s += part[c * 16 + tid];
        lr[row * Hh + tid] = (s > 20.f) ? s : log1pf(expf(s));
    }
}

// ---------------- depthwise causal conv-4 + residual (+ optional ReLU) ----------------
__global__ __launch_bounds__(256) void conv_kernel(const float* __restrict__ x,
                                                   const float* __restrict__ w,
                                                   float* __restrict__ y, int relu) {
    size_t idx = (size_t)blockIdx.x * 256 + threadIdx.x;
    if (idx >= (size_t)Bsz * Lsz * Dsz) return;
    int c = (int)(idx % Dsz);
    size_t bl = idx / Dsz;
    int t = (int)(bl % Lsz);
    size_t b = bl / Lsz;
    float acc = x[idx];   // residual
#pragma unroll
    for (int j = 0; j < KSZ; ++j) {
        int ts = t - (KSZ - 1) + j;
        if (ts >= 0) acc += w[c * KSZ + j] * x[(b * Lsz + ts) * Dsz + c];
    }
    if (relu) acc = fmaxf(acc, 0.f);
    y[idx] = acc;
}

// ---------------- sequential scan: one block per (b,h), state in LDS ----------------
__global__ __launch_bounds__(256) void scan_kernel(const float* __restrict__ q,
                                                   const float* __restrict__ k,
                                                   const float* __restrict__ v,
                                                   const float* __restrict__ lr,
                                                   float* __restrict__ o) {
    __shared__ float Wm[DKV * DKV];
    __shared__ float Hm[DKV * DKV];
    __shared__ float sq[DKV], sk[DKV], sv[DKV], ss[DKV];
    __shared__ float Tp[4][DKV], Yp[4][DKV], Op[4][DKV];
    __shared__ float red0[DKV], red1[DKV], red2[DKV];
    __shared__ float Dr[DKV];
    __shared__ float scal[4];   // 0: q.s, 1: |s|1, 2: lr_t, 3: q.q

    const int tid = threadIdx.x;
    const int bh  = blockIdx.x;
    const int b   = bh >> 4, h = bh & 15;

    for (int i = tid; i < DKV * DKV; i += 256) { Wm[i] = 0.f; Hm[i] = 0.f; }
    if (tid < DKV) ss[tid] = 0.f;

    const int r16 = tid >> 2;            // row for rank-1 updates
    const int c16 = (tid & 3) * 16;      // col base (16 elems/thread)
    const int j   = tid & 63;            // column for dot partials
    const int kc  = tid >> 6;            // 0..3 K-chunk

    const size_t basec = (size_t)b * Lsz * Dsz + (size_t)h * DKV;
    const size_t basel = (size_t)b * Lsz * Hh + h;

    for (int t = 0; t < Lsz; ++t) {
        __syncthreads();
        const size_t off = basec + (size_t)t * Dsz;
        if (tid < DKV) {
            sq[tid] = q[off + tid];
            sk[tid] = k[off + tid];
            sv[tid] = v[off + tid];
        } else if (tid == 64) {
            scal[2] = lr[basel + (size_t)t * Hh];
        }
        __syncthreads();

        // H += k v^T ; s += k ; scalar partials
        {
            float kr = sk[r16];
            float* hrow = &Hm[r16 * DKV + c16];
#pragma unroll
            for (int c = 0; c < 16; ++c) hrow[c] += kr * sv[c16 + c];
        }
        if (tid < DKV) {
            float snew = ss[tid] + sk[tid];
            ss[tid] = snew;
            float qv = sq[tid];
            red0[tid] = qv * snew;
            red1[tid] = fabsf(snew);
            red2[tid] = qv * qv;
        }
        __syncthreads();

        // T = q^T H, Y = q^T W partials; threads 64..66 also reduce scalars
        {
            float pT = 0.f, pY = 0.f;
            int kb = kc * 16;
#pragma unroll
            for (int kk = 0; kk < 16; ++kk) {
                float qv = sq[kb + kk];
                pT += qv * Hm[(kb + kk) * DKV + j];
                pY += qv * Wm[(kb + kk) * DKV + j];
            }
            Tp[kc][j] = pT; Yp[kc][j] = pY;
        }
        if (tid == 64)      { float a = 0.f; for (int i = 0; i < DKV; ++i) a += red0[i]; scal[0] = a; }
        else if (tid == 65) { float a = 0.f; for (int i = 0; i < DKV; ++i) a += red1[i]; scal[1] = a; }
        else if (tid == 66) { float a = 0.f; for (int i = 0; i < DKV; ++i) a += red2[i]; scal[3] = a; }
        __syncthreads();

        if (tid < DKV) {
            float Tv = Tp[0][tid] + Tp[1][tid] + Tp[2][tid] + Tp[3][tid];
            float Yv = Yp[0][tid] + Yp[1][tid] + Yp[2][tid] + Yp[3][tid];
            float alpha = scal[0] / (scal[1] + 1e-8f);
            float den   = scal[3] + 1e-6f;
            Dr[tid] = scal[2] * (Tv - alpha * Yv) / den;   // lr * Dres / den
        }
        __syncthreads();

        // W -= q (x) Dr
        {
            float qr = sq[r16];
            float* wrow = &Wm[r16 * DKV + c16];
#pragma unroll
            for (int c = 0; c < 16; ++c) wrow[c] -= qr * Dr[c16 + c];
        }
        __syncthreads();

        // o = q^T W partials
        {
            float pO = 0.f;
            int kb = kc * 16;
#pragma unroll
            for (int kk = 0; kk < 16; ++kk) pO += sq[kb + kk] * Wm[(kb + kk) * DKV + j];
            Op[kc][j] = pO;
        }
        __syncthreads();
        if (tid < DKV)
            o[off + tid] = Op[0][tid] + Op[1][tid] + Op[2][tid] + Op[3][tid];
    }
}

// ---------------- LayerNorm(64) * gate -> bf16 ----------------
__global__ __launch_bounds__(256) void ln_gate_kernel(const float* __restrict__ o,
                                                      const float* __restrict__ g,
                                                      const float* __restrict__ ln_g,
                                                      const float* __restrict__ ln_b,
                                                      unsigned short* __restrict__ out) {
    __shared__ float red[256];
    const int tid  = threadIdx.x;
    const int j    = tid & 63;
    const int base = tid & ~63;
    const size_t row = (size_t)blockIdx.x * 4 + (tid >> 6);
    const size_t idx = row * DKV + j;
    float val = o[idx];
    red[tid] = val; __syncthreads();
    for (int off = 32; off >= 1; off >>= 1) { if (j < off) red[tid] += red[tid + off]; __syncthreads(); }
    float mu = red[base] * (1.f / 64.f);
    __syncthreads();
    float d = val - mu;
    red[tid] = d * d; __syncthreads();
    for (int off = 32; off >= 1; off >>= 1) { if (j < off) red[tid] += red[tid + off]; __syncthreads(); }
    float var = red[base] * (1.f / 64.f);
    float y = d * rsqrtf(var + 1e-5f) * ln_g[j] + ln_b[j];
    out[idx] = f2bf(y * g[idx]);
}

// ---------------- host-side orchestration ----------------
extern "C" void kernel_launch(void* const* d_in, const int* in_sizes, int n_in,
                              void* d_out, int out_size, void* d_ws, size_t ws_size,
                              hipStream_t stream) {
    (void)in_sizes; (void)n_in; (void)out_size; (void)ws_size;
    const float* x      = (const float*)d_in[0];
    const float* Wq     = (const float*)d_in[1];
    const float* Wk     = (const float*)d_in[2];
    const float* Wv     = (const float*)d_in[3];
    const float* Wlr    = (const float*)d_in[4];
    const float* Wg     = (const float*)d_in[5];
    const float* Wo     = (const float*)d_in[6];
    const float* conv_q = (const float*)d_in[7];
    const float* conv_k = (const float*)d_in[8];
    const float* conv_v = (const float*)d_in[9];
    const float* ln_g   = (const float*)d_in[10];
    const float* ln_b   = (const float*)d_in[11];
    float* out = (float*)d_out;

    const size_t NEL = (size_t)Bsz * Lsz * Dsz;   // 16M
    const size_t WEL = (size_t)Dsz * Dsz;         // 1M
    const int M = Bsz * Lsz, N = Dsz, Kd = Dsz;

    char* ws = (char*)d_ws;
    unsigned short* xbf = (unsigned short*)ws;                 // 16M ush
    unsigned short* wqt = xbf + NEL;                           // 5 x 1M ush
    unsigned short* wkt = wqt + WEL;
    unsigned short* wvt = wkt + WEL;
    unsigned short* wgt = wvt + WEL;
    unsigned short* wot = wgt + WEL;
    float* fb = (float*)(ws + 44ull * 1024 * 1024);            // 16B-aligned float region
    float* Qp = fb;
    float* Kp = Qp + NEL;
    float* Vp = Kp + NEL;
    float* Gp = Vp + NEL;
    float* QC = Gp + NEL;
    float* KC = QC + NEL;
    float* VC = KC + NEL;
    float* LR = VC + NEL;                                      // B*L*H floats
    float* Op = Qp;                                            // reuse Q buffer
    unsigned short* OGBF = (unsigned short*)Kp;                // reuse K buffer

    // 1) x -> bf16
    cvt_bf16_kernel<<<dim3((unsigned)(NEL / 4 / 256)), dim3(256), 0, stream>>>(x, xbf, NEL);

    // 2) weights -> transposed bf16 (N x K)
    dim3 tb(32, 32), tg(Dsz / 32, Dsz / 32);
    transpose_cvt_kernel<<<tg, tb, 0, stream>>>(Wq, wqt, Kd, N);
    transpose_cvt_kernel<<<tg, tb, 0, stream>>>(Wk, wkt, Kd, N);
    transpose_cvt_kernel<<<tg, tb, 0, stream>>>(Wv, wvt, Kd, N);
    transpose_cvt_kernel<<<tg, tb, 0, stream>>>(Wg, wgt, Kd, N);
    transpose_cvt_kernel<<<tg, tb, 0, stream>>>(Wo, wot, Kd, N);

    // 3) projection GEMMs (WMMA bf16)
    dim3 gg(N / 128, M / 128), gb(256);
    gemm_bf16_wmma<<<gg, gb, 0, stream>>>(xbf, wqt, Qp, M, N, Kd);
    gemm_bf16_wmma<<<gg, gb, 0, stream>>>(xbf, wkt, Kp, M, N, Kd);
    gemm_bf16_wmma<<<gg, gb, 0, stream>>>(xbf, wvt, Vp, M, N, Kd);
    gemm_bf16_wmma<<<gg, gb, 0, stream>>>(xbf, wgt, Gp, M, N, Kd);

    // 4) learning rates
    lr_kernel<<<dim3(M), dim3(256), 0, stream>>>(x, Wlr, LR);

    // 5) depthwise conv + residual (+ReLU for q,k)
    unsigned cgrid = (unsigned)(NEL / 256);
    conv_kernel<<<dim3(cgrid), dim3(256), 0, stream>>>(Qp, conv_q, QC, 1);
    conv_kernel<<<dim3(cgrid), dim3(256), 0, stream>>>(Kp, conv_k, KC, 1);
    conv_kernel<<<dim3(cgrid), dim3(256), 0, stream>>>(Vp, conv_v, VC, 0);

    // 6) sequential scan, one block per (b,h)
    scan_kernel<<<dim3(Bsz * Hh), dim3(256), 0, stream>>>(QC, KC, VC, LR, Op);

    // 7) LayerNorm + gate, emit bf16
    unsigned rows = (unsigned)((size_t)Bsz * Lsz * Hh);
    ln_gate_kernel<<<dim3(rows / 4), dim3(256), 0, stream>>>(Op, Gp, ln_g, ln_b, OGBF);

    // 8) output projection GEMM (WMMA bf16) -> d_out (f32)
    gemm_bf16_wmma<<<gg, gb, 0, stream>>>(OGBF, wot, out, M, N, Kd);
}
